// GraphSAGEMassive_72232759984911
// MI455X (gfx1250) — matrix operand order repaired and checked by reading, not compiled, not run on hardware
//
#include <hip/hip_runtime.h>

// ---------------------------------------------------------------------------
// GraphSAGE 2-layer (mean aggr) for MI455X / gfx1250.
//  - Irregular phase: hardware global_atomic_add_f32 (inline CDNA5 asm,
//    device scope) into the L2-resident agg buffer (51 MB << 192 MB L2).
//  - Dense phase: v_wmma_f32_16x16x32_bf16; one wave owns a 16-row block and
//    computes ALL column tiles (A loaded once, 64 WMMAs/wave, 8 independent
//    accumulators), fused mean-divide / bias / relu / bf16-convert epilogues.
// ---------------------------------------------------------------------------

typedef __attribute__((ext_vector_type(16))) __bf16 v16bf;
typedef __attribute__((ext_vector_type(8)))  __bf16 v8bf;
typedef __attribute__((ext_vector_type(8)))  float  v8f;
typedef __attribute__((ext_vector_type(4)))  float  v4f;

#define NN 100000     // nodes (== 6250 * 16, exact tiles)
#define NE 1600000    // edges
#define DI 128
#define DH 128
#define DO 64
#define MT_TOT (NN / 16)   // 6250 row tiles

__device__ __forceinline__ void atomAddF(float* p, float v) {
  // CDNA5 VGLOBAL atomic, no return (STOREcnt-tracked; s_endpgm waits idle).
  asm volatile("global_atomic_add_f32 %0, %1, off scope:SCOPE_DEV"
               :: "v"((unsigned long long)(uintptr_t)p), "v"(v)
               : "memory");
}

// -------------------------------- utility ----------------------------------
__global__ void k_zero(float* __restrict__ p, long n) {
  long i = (long)blockIdx.x * blockDim.x + threadIdx.x;
  if (i < n) p[i] = 0.0f;
}

// Pack f32 weight matrix W[K][Nc] (row-major) into per-lane WMMA B-fragment
// order: tile t = nt*(K/32)+kc, lane L (half=L>>4, n=L&15), elem e in 0..15
// holds B[K = kc*32 + half*8 + e + (e>>3)*8][nt*16 + n] as bf16, contiguous per lane.
__global__ void k_pack_w(const float* __restrict__ W, __bf16* __restrict__ out,
                         int K, int Nc) {
  int tid = blockIdx.x * blockDim.x + threadIdx.x;
  if (tid >= K * Nc) return;
  int t = tid >> 9;
  int L = (tid >> 4) & 31;
  int e = tid & 15;
  int kTiles = K >> 5;
  int nt = t / kTiles;
  int kc = t - nt * kTiles;
  int half = L >> 4, n = L & 15;
  int k = kc * 32 + half * 8 + e + (e >> 3) * 8;
  out[tid] = (__bf16)W[k * Nc + nt * 16 + n];
}

// ------------------------------ irregular phase ----------------------------
__global__ void k_count(const long long* __restrict__ ei, float* __restrict__ cnt) {
  int i = blockIdx.x * blockDim.x + threadIdx.x;   // exactly NE threads
  atomAddF(&cnt[(int)ei[NE + i]], 1.0f);
}

// layer-0 scatter: 32 threads/edge, each gathers float4 of x[src], atomic-adds
__global__ void k_scatter_f32(const float* __restrict__ x,
                              const long long* __restrict__ ei,
                              float* __restrict__ agg) {
  int tid = blockIdx.x * blockDim.x + threadIdx.x;  // NE*32 threads exact
  int edge = tid >> 5;
  int c = (tid & 31) << 2;
  int s = (int)ei[edge];
  int d = (int)ei[NE + edge];
  v4f v = *(const v4f*)(x + (long)s * DI + c);
  float* p = agg + (long)d * DI + c;
  atomAddF(p + 0, v[0]);
  atomAddF(p + 1, v[1]);
  atomAddF(p + 2, v[2]);
  atomAddF(p + 3, v[3]);
}

// layer-1 scatter: 16 threads/edge, gathers 8 bf16 of h[src] (16B), f32 atomics
__global__ void k_scatter_bf16(const __bf16* __restrict__ h,
                               const long long* __restrict__ ei,
                               float* __restrict__ agg) {
  int tid = blockIdx.x * blockDim.x + threadIdx.x;  // NE*16 threads exact
  int edge = tid >> 4;
  int c = (tid & 15) << 3;
  int s = (int)ei[edge];
  int d = (int)ei[NE + edge];
  v8bf v = *(const v8bf*)(h + (long)s * DH + c);
  float* p = agg + (long)d * DH + c;
#pragma unroll
  for (int i = 0; i < 8; ++i) atomAddF(p + i, (float)v[i]);
}

// ------------------------------ WMMA GEMMs ---------------------------------
// A fragment (16-bit A 16x32 layout): lane m = lane&15, half = lane>>4.
// elems 0..7  = K (kc*32 + half*8 + 0..7)
// elems 8..15 = K (kc*32 + 16 + half*8 + 0..7)
__device__ __forceinline__ v16bf load_a_f32(const float* __restrict__ row,
                                            int o0, float scale) {
  v4f a0 = *(const v4f*)(row + o0);
  v4f a1 = *(const v4f*)(row + o0 + 4);
  v4f a2 = *(const v4f*)(row + o0 + 16);
  v4f a3 = *(const v4f*)(row + o0 + 20);
  v16bf A;
#pragma unroll
  for (int i = 0; i < 4; ++i) {
    A[i]      = (__bf16)(a0[i] * scale);
    A[4 + i]  = (__bf16)(a1[i] * scale);
    A[8 + i]  = (__bf16)(a2[i] * scale);
    A[12 + i] = (__bf16)(a3[i] * scale);
  }
  return A;
}

__device__ __forceinline__ v16bf load_a_bf16(const __bf16* __restrict__ row, int o0) {
  v8bf lo = *(const v8bf*)(row + o0);
  v8bf hi = *(const v8bf*)(row + o0 + 16);
  v16bf A;
#pragma unroll
  for (int i = 0; i < 8; ++i) { A[i] = lo[i]; A[8 + i] = hi[i]; }
  return A;
}

// Layer 0: h = relu(mean @ Wl0 + b + x @ Wr0), h stored bf16.
// One wave per 16-row block computes all 8 column tiles (128 cols).
__global__ __launch_bounds__(256) void k_gemm0(
    const float* __restrict__ agg, const float* __restrict__ cnt,
    const float* __restrict__ x, const __bf16* __restrict__ Wlp,
    const __bf16* __restrict__ Wrp, const float* __restrict__ bias,
    __bf16* __restrict__ h) {
  const int wave = threadIdx.x >> 5;
  const int lane = threadIdx.x & 31;
  const int mt   = blockIdx.x * 8 + wave;   // row tile owned by this wave
  if (mt >= MT_TOT) return;                 // wave-uniform guard (EXEC all-1s)
  const int half = lane >> 4;
  const int row  = mt * 16 + (lane & 15);
  const float inv = 1.0f / fmaxf(cnt[row], 1.0f);
  const float* aggRow = agg + (long)row * DI;
  const float* xRow   = x   + (long)row * DI;
  v8f acc[8] = {};
#pragma unroll
  for (int kc = 0; kc < 4; ++kc) {
    const int o0 = kc * 32 + half * 8;
    v16bf Am = load_a_f32(aggRow, o0, inv);    // mean fragment (fused divide)
    v16bf Ar = load_a_f32(xRow,  o0, 1.0f);    // root fragment
#pragma unroll
    for (int nt = 0; nt < 8; ++nt) {
      v16bf Bl = *(const v16bf*)(Wlp + (((nt << 2) + kc) * 32 + lane) * 16);
      v16bf Br = *(const v16bf*)(Wrp + (((nt << 2) + kc) * 32 + lane) * 16);
      acc[nt] = __builtin_amdgcn_wmma_f32_16x16x32_bf16(
          false, Am, false, Bl, (short)0, acc[nt], false, false);
      acc[nt] = __builtin_amdgcn_wmma_f32_16x16x32_bf16(
          false, Ar, false, Br, (short)0, acc[nt], false, false);
    }
  }
  const int n = lane & 15;
#pragma unroll
  for (int nt = 0; nt < 8; ++nt) {
    const int col = nt * 16 + n;
    const float b = bias[col];
#pragma unroll
    for (int r = 0; r < 8; ++r) {
      float v = fmaxf(acc[nt][r] + b, 0.0f);                       // bias+relu
      h[(long)(mt * 16 + r + half * 8) * DH + col] = (__bf16)v;    // C layout
    }
  }
}

// Layer 1: out = mean1 @ Wl1 + b + h @ Wr1, f32 out (64 cols -> 4 tiles/wave).
__global__ __launch_bounds__(256) void k_gemm1(
    const float* __restrict__ agg, const float* __restrict__ cnt,
    const __bf16* __restrict__ hbf, const __bf16* __restrict__ Wlp,
    const __bf16* __restrict__ Wrp, const float* __restrict__ bias,
    float* __restrict__ out) {
  const int wave = threadIdx.x >> 5;
  const int lane = threadIdx.x & 31;
  const int mt   = blockIdx.x * 8 + wave;
  if (mt >= MT_TOT) return;
  const int half = lane >> 4;
  const int row  = mt * 16 + (lane & 15);
  const float inv = 1.0f / fmaxf(cnt[row], 1.0f);
  const float*  aggRow = agg + (long)row * DH;
  const __bf16* hRow   = hbf + (long)row * DH;
  v8f acc[4] = {};
#pragma unroll
  for (int kc = 0; kc < 4; ++kc) {
    const int o0 = kc * 32 + half * 8;
    v16bf Am = load_a_f32(aggRow, o0, inv);
    v16bf Ar = load_a_bf16(hRow, o0);
#pragma unroll
    for (int nt = 0; nt < 4; ++nt) {
      v16bf Bl = *(const v16bf*)(Wlp + (((nt << 2) + kc) * 32 + lane) * 16);
      v16bf Br = *(const v16bf*)(Wrp + (((nt << 2) + kc) * 32 + lane) * 16);
      acc[nt] = __builtin_amdgcn_wmma_f32_16x16x32_bf16(
          false, Am, false, Bl, (short)0, acc[nt], false, false);
      acc[nt] = __builtin_amdgcn_wmma_f32_16x16x32_bf16(
          false, Ar, false, Br, (short)0, acc[nt], false, false);
    }
  }
  const int n = lane & 15;
#pragma unroll
  for (int nt = 0; nt < 4; ++nt) {
    const int col = nt * 16 + n;
    const float b = bias[col];
#pragma unroll
    for (int r = 0; r < 8; ++r)
      out[(long)(mt * 16 + r + half * 8) * DO + col] = acc[nt][r] + b;
  }
}

// ------------------------------- launcher ----------------------------------
extern "C" void kernel_launch(void* const* d_in, const int* in_sizes, int n_in,
                              void* d_out, int out_size, void* d_ws, size_t ws_size,
                              hipStream_t stream) {
  const float*     x    = (const float*)d_in[0];
  const long long* ei   = (const long long*)d_in[1];   // int64 edge_index [2,E]
  const float*     Wl0  = (const float*)d_in[2];
  const float*     bl0  = (const float*)d_in[3];
  const float*     Wr0  = (const float*)d_in[4];
  const float*     Wl1  = (const float*)d_in[5];
  const float*     bl1  = (const float*)d_in[6];
  const float*     Wr1  = (const float*)d_in[7];

  // workspace partition (~77.3 MB, 256B-aligned slabs)
  char*  base = (char*)d_ws;
  size_t off  = 0;
  float* agg  = (float*)(base + off); off += (size_t)NN * DI * 4;        // 51.2 MB
  float* cnt  = (float*)(base + off); off += (((size_t)NN * 4 + 255) / 256) * 256;
  __bf16* hbf = (__bf16*)(base + off); off += (size_t)NN * DH * 2;       // 25.6 MB
  __bf16* Wl0p = (__bf16*)(base + off); off += (size_t)DI * DH * 2;
  __bf16* Wr0p = (__bf16*)(base + off); off += (size_t)DI * DH * 2;
  __bf16* Wl1p = (__bf16*)(base + off); off += (size_t)DH * DO * 2;
  __bf16* Wr1p = (__bf16*)(base + off); off += (size_t)DH * DO * 2;

  const long aggN = (long)NN * DI;
  const int  gemmGrid = (MT_TOT + 7) / 8;   // 8 waves (row tiles) per block

  // zero accumulators
  k_zero<<<(int)(aggN / 256), 256, 0, stream>>>(agg, aggN);
  k_zero<<<(NN + 255) / 256, 256, 0, stream>>>(cnt, NN);

  // pack weights into WMMA B-fragment order (bf16)
  k_pack_w<<<(DI * DH) / 256, 256, 0, stream>>>(Wl0, Wl0p, DI, DH);
  k_pack_w<<<(DI * DH) / 256, 256, 0, stream>>>(Wr0, Wr0p, DI, DH);
  k_pack_w<<<(DH * DO) / 256, 256, 0, stream>>>(Wl1, Wl1p, DH, DO);
  k_pack_w<<<(DH * DO) / 256, 256, 0, stream>>>(Wr1, Wr1p, DH, DO);

  // degrees (shared by both layers)
  k_count<<<NE / 256, 256, 0, stream>>>(ei, cnt);

  // ---- layer 0 ----
  k_scatter_f32<<<(NE / 8), 256, 0, stream>>>(x, ei, agg);        // NE*32 thr
  k_gemm0<<<gemmGrid, 256, 0, stream>>>(agg, cnt, x, Wl0p, Wr0p, bl0, hbf);

  // ---- layer 1 ----
  k_zero<<<(int)(aggN / 256), 256, 0, stream>>>(agg, aggN);
  k_scatter_bf16<<<(NE / 16), 256, 0, stream>>>(hbf, ei, agg);    // NE*16 thr
  k_gemm1<<<gemmGrid, 256, 0, stream>>>(agg, cnt, hbf, Wl1p, Wr1p, bl1,
                                        (float*)d_out);
}
